// EdgeGNN_38654705664133
// MI455X (gfx1250) — matrix operand order repaired and checked by reading, not compiled, not run on hardware
//
#include <hip/hip_runtime.h>
#include <hip/hip_bf16.h>

// ---------------------------------------------------------------------------
// CDNA5 / gfx1250 EdgeGNN pipeline.
// GEMMs use v_wmma_f32_16x16x32_bf16 (wave32, fp32 accumulate), with the
// 16-row A stripe preloaded into registers once and reused across all
// column tiles (A traffic = 1x instead of NC/16 x).
// Scatter uses global_atomic_add_f32. Head uses wave32 shfl_xor reduction.
// ---------------------------------------------------------------------------

typedef __attribute__((ext_vector_type(16))) __bf16 v16bf;
typedef __attribute__((ext_vector_type(8)))  float  v8f;

__device__ __forceinline__ float leaky(float v) { return v > 0.f ? v : 0.01f * v; }

// D[nrows, NC] = act(A[nrows, K] @ B[K, NC] (+ bias))
// One wave (32 threads) computes a full 16 x NC output stripe.
// Fragment layouts per CDNA5 ISA 7.12.2:
//   A 16x32 bf16: lane m = l&15, half = l>>4; elem j -> K = (j<8 ? 8*half+j : 16+8*half+(j-8))
//   B 32x16 bf16: lane n = l&15, half = l>>4; elem j -> K = (j<8 ? 8*half+j : 16+8*half+(j-8))
//   C/D f32 16x16: lane n = l&15; vgpr r -> M = r + 8*(l>>4)
template <int K, int NC, bool ACT, bool BIAS>
__global__ void __launch_bounds__(32)
wmma_gemm(const float* __restrict__ A, const float* __restrict__ B,
          const float* __restrict__ bias, float* __restrict__ D, int nrows)
{
    constexpr int KT = K / 32;    // k-steps
    constexpr int NT = NC / 16;   // column tiles

    const int tm   = blockIdx.x;          // 16-row stripe
    const int lane = threadIdx.x;         // 0..31 (wave32)
    const int m    = lane & 15;
    const int half = lane >> 4;

    // ---- preload the whole 16 x K A stripe as bf16 fragments ----
    const float* arow = A + (size_t)(tm * 16 + m) * K;
    v16bf afrag[KT];
#pragma unroll
    for (int ks = 0; ks < KT; ++ks) {
        const float* p0 = arow + ks * 32 + 8 * half;        // 8 contiguous
        const float* p1 = arow + ks * 32 + 16 + 8 * half;   // 8 contiguous
#pragma unroll
        for (int j = 0; j < 8; ++j) {
            afrag[ks][j]     = (__bf16)p0[j];
            afrag[ks][8 + j] = (__bf16)p1[j];
        }
    }

    const bool fullTile = (tm * 16 + 16) <= nrows;

    // ---- loop over the 16-column tiles, reusing A fragments ----
#pragma unroll
    for (int tn = 0; tn < NT; ++tn) {
        const int n = tn * 16 + m;
        v8f acc = {};
#pragma unroll
        for (int ks = 0; ks < KT; ++ks) {
            // B: column n, K rows ks*32 + (j<8 ? 8*half+j : 16+8*half+(j-8))
            const float* q = B + (size_t)(ks * 32) * NC + n;
            v16bf b;
#pragma unroll
            for (int j = 0; j < 16; ++j)
                b[j] = (__bf16)q[(size_t)(j < 8 ? 8 * half + j : 16 + 8 * half + (j - 8)) * NC];
            acc = __builtin_amdgcn_wmma_f32_16x16x32_bf16(
                /*neg_a=*/false, afrag[ks], /*neg_b=*/false, b,
                /*c_mod=*/(short)0, acc, /*reuse_a=*/false, /*reuse_b=*/false);
        }

        const float bv = BIAS ? bias[n] : 0.f;
        if (fullTile) {
#pragma unroll
            for (int r = 0; r < 8; ++r) {
                const int mm = tm * 16 + r + 8 * half;
                float v = acc[r] + bv;
                if (ACT) v = leaky(v);
                D[(size_t)mm * NC + n] = v;
            }
        } else {
#pragma unroll
            for (int r = 0; r < 8; ++r) {
                const int mm = tm * 16 + r + 8 * half;
                if (mm < nrows) {
                    float v = acc[r] + bv;
                    if (ACT) v = leaky(v);
                    D[(size_t)mm * NC + n] = v;
                }
            }
        }
    }
}

// deg[i] = 1.0 (self loop)
__global__ void deg_init(float* __restrict__ deg, int N)
{
    int i = blockIdx.x * 256 + threadIdx.x;
    if (i < N) deg[i] = 1.f;
}

// deg[col[e]] += 1
__global__ void deg_count(const int* __restrict__ col, float* __restrict__ deg, int E)
{
    int e = blockIdx.x * 256 + threadIdx.x;
    if (e < E) atomicAdd(&deg[col[e]], 1.f);
}

// dis[i] = rsqrt(deg[i])
__global__ void rsqrt_k(const float* __restrict__ deg, float* __restrict__ dis, int N)
{
    int i = blockIdx.x * 256 + threadIdx.x;
    if (i < N) dis[i] = rsqrtf(deg[i]);
}

// agg[i, 0:64] = xw[i, 0:64] * dis[i]^2  (self-loop term; also initializes agg)
// one thread per 4 floats (float4 path, b128 loads/stores)
__global__ void agg_selfinit(const float4* __restrict__ xw, const float* __restrict__ dis,
                             float4* __restrict__ agg, int total4)
{
    int i = blockIdx.x * 256 + threadIdx.x;
    if (i < total4) {
        float s = dis[i >> 4];          // 16 float4 per node row
        float w = s * s;
        float4 v = xw[i];
        v.x *= w; v.y *= w; v.z *= w; v.w *= w;
        agg[i] = v;
    }
}

// one wave per edge: agg[col] += dis[row]*dis[col] * xw[row]   (64 floats/row)
__global__ void __launch_bounds__(256)
gcn_scatter(const float* __restrict__ xw, const int* __restrict__ row,
            const int* __restrict__ col, const float* __restrict__ dis,
            float* __restrict__ agg, int E)
{
    int wave = (blockIdx.x * 256 + threadIdx.x) >> 5;
    int lane = threadIdx.x & 31;
    if (wave >= E) return;
    const int r = row[wave];
    const int c = col[wave];
    const float norm = dis[r] * dis[c];
    const float* src = xw + (size_t)r * 64;
    float* dst = agg + (size_t)c * 64;
    atomicAdd(&dst[lane],      norm * src[lane]);
    atomicAdd(&dst[lane + 32], norm * src[lane + 32]);
}

// h[i,f] = leaky(agg[i,f] + b[f]) ; float4 path
__global__ void gcn_finalize(const float4* __restrict__ agg, const float* __restrict__ b,
                             float4* __restrict__ h, int total4)
{
    int i = blockIdx.x * 256 + threadIdx.x;
    if (i < total4) {
        int f = (i & 15) * 4;           // feature offset within 64
        float4 v = agg[i];
        v.x = leaky(v.x + b[f]);
        v.y = leaky(v.y + b[f + 1]);
        v.z = leaky(v.z + b[f + 2]);
        v.w = leaky(v.w + b[f + 3]);
        h[i] = v;
    }
}

// one wave per label edge: out[e] = [h_src | h_dst | ea] . Wp + bp
__global__ void __launch_bounds__(256)
edge_score(const float* __restrict__ h, const int* __restrict__ eli,
           const float* __restrict__ ea, const float* __restrict__ Wp,
           const float* __restrict__ bp, float* __restrict__ out, int EL)
{
    int wave = (blockIdx.x * 256 + threadIdx.x) >> 5;
    int lane = threadIdx.x & 31;
    if (wave >= EL) return;
    const int s = eli[wave];
    const int d = eli[EL + wave];
    const float* hs = h + (size_t)s * 64;
    const float* hd = h + (size_t)d * 64;
    float acc = Wp[lane]      * hs[lane]
              + Wp[lane + 32] * hs[lane + 32]
              + Wp[lane + 64] * hd[lane]
              + Wp[lane + 96] * hd[lane + 32];
    if (lane < 8) acc += Wp[128 + lane] * ea[(size_t)wave * 8 + lane];
    // wave32 butterfly reduction
#pragma unroll
    for (int off = 16; off > 0; off >>= 1)
        acc += __shfl_xor(acc, off, 32);
    if (lane == 0) out[wave] = acc + bp[0];
}

extern "C" void kernel_launch(void* const* d_in, const int* in_sizes, int n_in,
                              void* d_out, int out_size, void* d_ws, size_t ws_size,
                              hipStream_t stream)
{
    const float* x   = (const float*)d_in[0];
    const int*   ei  = (const int*)d_in[1];
    const int*   eli = (const int*)d_in[2];
    const float* ea  = (const float*)d_in[3];
    const float* W1  = (const float*)d_in[4];
    const float* b1  = (const float*)d_in[5];
    const float* W2  = (const float*)d_in[6];
    const float* b2  = (const float*)d_in[7];
    const float* Wg1 = (const float*)d_in[8];
    const float* bg1 = (const float*)d_in[9];
    const float* Wg2 = (const float*)d_in[10];
    const float* bg2 = (const float*)d_in[11];
    const float* Wp  = (const float*)d_in[12];
    const float* bp  = (const float*)d_in[13];

    const int N  = in_sizes[0] / 256;   // 100000
    const int E  = in_sizes[1] / 2;     // 1600000
    const int EL = in_sizes[2] / 2;     // 200000

    const int* row = ei;
    const int* col = ei + E;

    // workspace layout (floats)
    float* ws  = (float*)d_ws;
    float* h1  = ws;                        // [N,128] MLP hidden
    float* hm  = h1 + (size_t)N * 128;      // [N,64]  MLP out -> reused for GCN1 out
    float* xw  = hm + (size_t)N * 64;       // [N,64]  h @ Wg
    float* agg = xw + (size_t)N * 64;       // [N,64]  aggregation buffer
    float* deg = agg + (size_t)N * 64;      // [N]
    float* dis = deg + N;                   // [N]

    float* out_scores = (float*)d_out;      // [EL]
    float* h_out      = out_scores + EL;    // [N,64] final embeddings

    const int rowTiles  = (N + 15) / 16;
    const int total4    = N * 16;                 // N*64/4 float4 elements
    const int vec4Blk   = (total4 + 255) / 256;
    const int nBlk      = (N + 255) / 256;
    const int eBlk      = (E + 255) / 256;
    const int eWaveBlk  = (E * 32 + 255) / 256;   // one wave per edge
    const int elWaveBlk = (EL * 32 + 255) / 256;  // one wave per label edge

    // ---- preprocess MLP: h1 = leaky(x@W1+b1); hm = leaky(h1@W2+b2) ----
    wmma_gemm<256, 128, true, true><<<rowTiles, 32, 0, stream>>>(x, W1, b1, h1, N);
    wmma_gemm<128, 64,  true, true><<<rowTiles, 32, 0, stream>>>(h1, W2, b2, hm, N);

    // ---- degree / normalization (shared by both GCN layers) ----
    deg_init<<<nBlk, 256, 0, stream>>>(deg, N);
    deg_count<<<eBlk, 256, 0, stream>>>(col, deg, E);
    rsqrt_k<<<nBlk, 256, 0, stream>>>(deg, dis, N);

    // ---- GCN layer 1 ----
    wmma_gemm<64, 64, false, false><<<rowTiles, 32, 0, stream>>>(hm, Wg1, nullptr, xw, N);
    agg_selfinit<<<vec4Blk, 256, 0, stream>>>((const float4*)xw, dis, (float4*)agg, total4);
    gcn_scatter<<<eWaveBlk, 256, 0, stream>>>(xw, row, col, dis, agg, E);
    gcn_finalize<<<vec4Blk, 256, 0, stream>>>((const float4*)agg, bg1, (float4*)hm, total4);

    // ---- GCN layer 2 (finalize writes directly into d_out h-region) ----
    wmma_gemm<64, 64, false, false><<<rowTiles, 32, 0, stream>>>(hm, Wg2, nullptr, xw, N);
    agg_selfinit<<<vec4Blk, 256, 0, stream>>>((const float4*)xw, dis, (float4*)agg, total4);
    gcn_scatter<<<eWaveBlk, 256, 0, stream>>>(xw, row, col, dis, agg, E);
    gcn_finalize<<<vec4Blk, 256, 0, stream>>>((const float4*)agg, bg2, (float4*)h_out, total4);

    // ---- edge scoring head ----
    edge_score<<<elWaveBlk, 256, 0, stream>>>(h_out, eli, ea, Wp, bp, out_scores, EL);
}
